// LSTMModel_86689619902598
// MI455X (gfx1250) — matrix operand order repaired and checked by reading, not compiled, not run on hardware
//
#include <hip/hip_runtime.h>

typedef __attribute__((ext_vector_type(16))) _Float16 v16h;
typedef __attribute__((ext_vector_type(8)))  _Float16 v8h;
typedef __attribute__((ext_vector_type(8)))  float    v8f;

#define LEAKY(x) ((x) > 0.0f ? (x) : 0.01f * (x))

// ---- gfx1250 async global->LDS staging (ASYNCcnt) ----
#if defined(__has_builtin)
#if __has_builtin(__builtin_amdgcn_global_load_async_to_lds_b128) && \
    __has_builtin(__builtin_amdgcn_global_load_async_to_lds_b32)
#define ASYNC_LDS 1
#endif
#endif

#if defined(ASYNC_LDS)
#if __has_builtin(__builtin_amdgcn_s_wait_asynccnt)
#define WAIT_ASYNC() __builtin_amdgcn_s_wait_asynccnt(0)
#else
#define WAIT_ASYNC() asm volatile("s_wait_asynccnt 0x0" ::: "memory")
#endif
typedef int vi4 __attribute__((vector_size(16)));
#define ASG __attribute__((address_space(1)))   // global
#define ASL __attribute__((address_space(3)))   // LDS
#endif

// ---------- WMMA fragment helpers (CDNA5 16x16x32 f16, wave32) ----------
// A (16x32 MxK) from row-major A[M][K]; per-lane data = two contiguous 16B chunks.
__device__ __forceinline__ v16h load_a_vec(const _Float16* A, int lda, int row0, int k0) {
  int lane = threadIdx.x & 31;
  int l = lane & 15, hi = lane >> 4;
  const _Float16* base = A + (size_t)(row0 + l) * lda + k0 + hi * 16;
  v8h lo = *(const v8h*)base;          // K = hi*16 + 0..7
  v8h hiv = *(const v8h*)(base + 8);   // K = hi*16 + 8..15
  return __builtin_shufflevector(lo, hiv, 0, 1, 2, 3, 4, 5, 6, 7,
                                 8, 9, 10, 11, 12, 13, 14, 15);
}

// B (32x16 KxN) where B[k][n] = Wt[n][k]; per-lane data = 32 contiguous bytes.
__device__ __forceinline__ v16h load_b_vec(const _Float16* Wt, int ldb, int n0, int k0) {
  int lane = threadIdx.x & 31;
  int l = lane & 15, hi = lane >> 4;
  return *(const v16h*)(Wt + (size_t)(n0 + l) * ldb + k0 + hi * 16);
}

__device__ __forceinline__ v8f wmma_f16(v16h a, v16h b, v8f c) {
  return __builtin_amdgcn_wmma_f32_16x16x32_f16(false, a, false, b, (short)0, c, false, false);
}

// ---------- problem constants ----------
#define BB    32
#define TT    64
#define LL    1500
#define HH    128
#define DCC   100
#define NBT   2048    // B*T
#define P2    165     // conv2 out len
#define P3    54      // conv3 out len
#define FLATN 1728    // 32*54

// ---------- kernel 0: weight conversion (+ lin_w K-permute) + feats zero ----------
__global__ void prep_kernel(const float* __restrict__ w2, const float* __restrict__ w3,
                            const float* __restrict__ lin,
                            const float* __restrict__ wih0, const float* __restrict__ whh0,
                            const float* __restrict__ wih1, const float* __restrict__ whh1,
                            _Float16* __restrict__ w2h, _Float16* __restrict__ w3h,
                            _Float16* __restrict__ linh,
                            _Float16* __restrict__ wih0h, _Float16* __restrict__ whh0h,
                            _Float16* __restrict__ wih1h, _Float16* __restrict__ whh1h,
                            _Float16* __restrict__ feats) {
  int tid = blockIdx.x * blockDim.x + threadIdx.x;
  int np = gridDim.x * blockDim.x;
  for (int i = tid; i < 32 * 640; i += np) w2h[i] = (_Float16)w2[i];
  for (int i = tid; i < 32 * 160; i += np) w3h[i] = (_Float16)w3[i];
  // permute lin_w K ordering: k' = lpos*32 + oc  <->  flat = oc*54 + lpos
  for (int i = tid; i < DCC * FLATN; i += np) {
    int d = i / FLATN, kp = i - d * FLATN;
    int lpos = kp >> 5, oc = kp & 31;
    linh[i] = (_Float16)lin[(size_t)d * FLATN + oc * P3 + lpos];
  }
  for (int i = tid; i < 512 * 128; i += np) {
    int g = i >> 7, k = i & 127;
    wih0h[i] = (k < DCC) ? (_Float16)wih0[g * DCC + k] : (_Float16)0.0f;  // pad K 100->128
  }
  for (int i = tid; i < 512 * 128; i += np) whh0h[i] = (_Float16)whh0[i];
  for (int i = tid; i < 512 * 128; i += np) wih1h[i] = (_Float16)wih1[i];
  for (int i = tid; i < 512 * 128; i += np) whh1h[i] = (_Float16)whh1[i];
  for (int i = tid; i < TT * BB * 128; i += np) feats[i] = (_Float16)0.0f;  // d>=100 pad
}

// ---------- kernel 1: fused conv1 + conv2 (WMMA) ----------
// grid (2048, 6), block 128 (4 waves). One row n, 32 conv2 positions per block.
// a2 layout: [n][p][oc]  (position-major so C tiles store as contiguous v8h)
__global__ void conv12_kernel(const float* __restrict__ X, const float* __restrict__ w1,
                              const _Float16* __restrict__ w2, _Float16* __restrict__ a2) {
  __shared__ alignas(16) float Xwin[296];
  __shared__ alignas(32) _Float16 a1T[32 * 640];  // [subpos][kidx=ic*5+kk]
  int n = blockIdx.x;
  int P0 = blockIdx.y * 32;
  int xbase = 9 * P0;
#if defined(ASYNC_LDS)
  for (int i = threadIdx.x; i < 296; i += 128) {
    int gi = xbase + i;
    if (gi > LL - 1) gi = LL - 1;
    __builtin_amdgcn_global_load_async_to_lds_b32(
        (ASG int*)(X + (size_t)n * LL + gi), (ASL int*)&Xwin[i], 0, 0);
  }
  WAIT_ASYNC();
#else
  for (int i = threadIdx.x; i < 296; i += 128) {
    int gi = xbase + i;
    if (gi > LL - 1) gi = LL - 1;
    Xwin[i] = X[(size_t)n * LL + gi];
  }
#endif
  __syncthreads();
  {
    int ic = threadIdx.x;  // 0..127
    float w[5];
#pragma unroll
    for (int j = 0; j < 5; ++j) w[j] = w1[ic * 5 + j];
    for (int q = 0; q < 98; ++q) {   // q = 3*subpos + kk
      float s = 0.0f;
#pragma unroll
      for (int j = 0; j < 5; ++j) s += w[j] * Xwin[3 * q + j];
      _Float16 hv = (_Float16)LEAKY(s);
      int kk = q % 3;                // kk in {0,1,2}
      int sp = q / 3;
      if (sp < 32) a1T[sp * 640 + ic * 5 + kk] = hv;
      if (kk + 3 < 5 && sp >= 1)     // second tap alias: kk+3 in {3,4}
        a1T[(sp - 1) * 640 + ic * 5 + kk + 3] = hv;
    }
  }
  __syncthreads();
  int wave = threadIdx.x >> 5;
  int lane = threadIdx.x & 31;
  int l = lane & 15, hi = lane >> 4;
  int mt = wave >> 1;        // oc tile 0..1
  int ch = wave & 1;         // position half-tile
  v8f acc = {};
  for (int ks = 0; ks < 20; ++ks) {  // K = 640
    int k0 = ks * 32;
    v16h a = load_a_vec(w2, 640, mt * 16, k0);
    v16h b = load_b_vec(a1T, 640, ch * 16, k0);
    acc = wmma_f16(a, b, acc);
  }
  int p = P0 + ch * 16 + l;
  if (p < P2) {
    v8h o;
#pragma unroll
    for (int r = 0; r < 8; ++r) o[r] = (_Float16)LEAKY(acc[r]);
    *(v8h*)(a2 + (size_t)n * 32 * P2 + p * 32 + mt * 16 + hi * 8) = o;
  }
}

// ---------- kernel 2: conv3 (WMMA) ----------
// grid 2048, block 256 (8 waves): 2 oc-tiles x 4 pos-tiles (54 padded to 64)
// a3 layout: [n][lpos][oc]  (K ordering lpos*32+oc, matches permuted lin_w)
__global__ void conv3_kernel(const _Float16* __restrict__ a2, const _Float16* __restrict__ w3,
                             _Float16* __restrict__ a3) {
  __shared__ alignas(16) _Float16 a2win[P2 * 32];   // [p][ic]
  __shared__ alignas(32) _Float16 a2T[64 * 160];    // [lpos][kidx=ic*5+kk]
  int n = blockIdx.x;
#if defined(ASYNC_LDS)
  for (int i = threadIdx.x; i < P2 * 32 / 8; i += 256) {   // 660 x 16B chunks
    __builtin_amdgcn_global_load_async_to_lds_b128(
        (ASG vi4*)(a2 + (size_t)n * 32 * P2 + i * 8), (ASL vi4*)&a2win[i * 8], 0, 0);
  }
  WAIT_ASYNC();
#else
  for (int i = threadIdx.x; i < P2 * 32; i += 256)
    a2win[i] = a2[(size_t)n * 32 * P2 + i];
#endif
  __syncthreads();
  for (int i = threadIdx.x; i < 64 * 160; i += 256) {
    int lpos = i / 160, kidx = i - lpos * 160;
    int icx = kidx / 5, kk = kidx - icx * 5;
    int q = 3 * lpos + kk;
    if (q > P2 - 1) q = P2 - 1;   // clamp for padded positions (never stored)
    a2T[i] = a2win[q * 32 + icx];
  }
  __syncthreads();
  int wave = threadIdx.x >> 5;
  int lane = threadIdx.x & 31;
  int l = lane & 15, hi = lane >> 4;
  int mt = wave >> 2;           // 0..1
  int pt = wave & 3;            // 0..3
  v8f acc = {};
  for (int ks = 0; ks < 5; ++ks) {  // K = 160
    int k0 = ks * 32;
    v16h a = load_a_vec(w3, 160, mt * 16, k0);
    v16h b = load_b_vec(a2T, 160, pt * 16, k0);
    acc = wmma_f16(a, b, acc);
  }
  int lpos = pt * 16 + l;
  if (lpos < P3) {
    v8h o;
#pragma unroll
    for (int r = 0; r < 8; ++r) o[r] = (_Float16)LEAKY(acc[r]);
    *(v8h*)(a3 + (size_t)n * FLATN + lpos * 32 + mt * 16 + hi * 8) = o;
  }
}

// ---------- kernel 3: linear (2048x1728)x(1728x100), output time-major padded ----------
// grid (128, 7), block 32 (one wave per 16x16 tile)
__global__ void linear_kernel(const _Float16* __restrict__ a3, const _Float16* __restrict__ linh,
                              _Float16* __restrict__ feats) {
  int lane = threadIdx.x & 31;
  int l = lane & 15, hi = lane >> 4;
  int m0 = blockIdx.x * 16, n0 = blockIdx.y * 16;
  int d = n0 + l;
  v8f acc = {};
  for (int ks = 0; ks < 54; ++ks) {  // K = 1728 (permuted ordering on both operands)
    int k0 = ks * 32;
    v16h a = load_a_vec(a3, FLATN, m0, k0);
    v16h b = {};
    if (d < DCC)
      b = *(const v16h*)(linh + (size_t)d * FLATN + k0 + hi * 16);
    acc = wmma_f16(a, b, acc);
  }
#pragma unroll
  for (int r = 0; r < 8; ++r) {
    int m = m0 + hi * 8 + r;  // row n = b*64 + t
    int bb = m >> 6;
    int t = m & 63;
    if (d < DCC)
      feats[((size_t)t * BB + bb) * 128 + d] = (_Float16)acc[r];
  }
}

// ---------- kernel 4: persistent 2-layer LSTM + FC head ----------
// gates LDS is gate-major [g(512)][b(32)] so each lane's C column stores as one v8f.
__device__ __forceinline__ void lstm_gemm(const _Float16* x, const _Float16* h,
                                          const _Float16* wih, const _Float16* whh,
                                          float* gates) {
  int wave = threadIdx.x >> 5;
  int lane = threadIdx.x & 31;
  int l = lane & 15, hi = lane >> 4;
  for (int it = 0; it < 4; ++it) {
    int tile = wave * 4 + it;     // 64 tiles: 2 (batch) x 32 (gate cols)
    int m0 = (tile >> 5) * 16;
    int n0 = (tile & 31) * 16;
    v8f acc = {};
#pragma unroll
    for (int ks = 0; ks < 4; ++ks) {
      v16h a = load_a_vec(x, 128, m0, ks * 32);
      v16h b = load_b_vec(wih, 128, n0, ks * 32);
      acc = wmma_f16(a, b, acc);
    }
#pragma unroll
    for (int ks = 0; ks < 4; ++ks) {
      v16h a = load_a_vec(h, 128, m0, ks * 32);
      v16h b = load_b_vec(whh, 128, n0, ks * 32);
      acc = wmma_f16(a, b, acc);
    }
    *(v8f*)(gates + (n0 + l) * 32 + m0 + hi * 8) = acc;
  }
}

__device__ __forceinline__ void lstm_act(const float* gates, float* c, _Float16* h,
                                         float* hf, const int* lengths, int t) {
  for (int idx = threadIdx.x; idx < BB * HH; idx += blockDim.x) {
    int b = idx >> 7, j = idx & 127;
    if (t < lengths[b]) {     // masked update: h,c unchanged when t >= length
      float ig = gates[j * 32 + b];
      float fg = gates[(128 + j) * 32 + b];
      float gg = gates[(256 + j) * 32 + b];
      float og = gates[(384 + j) * 32 + b];
      float is = 1.0f / (1.0f + __expf(-ig));
      float fs = 1.0f / (1.0f + __expf(-fg));
      float os = 1.0f / (1.0f + __expf(-og));
      float gt = tanhf(gg);
      float cn = fs * c[idx] + is * gt;
      float hn = os * tanhf(cn);
      c[idx] = cn;
      h[idx] = (_Float16)hn;
      if (hf) hf[idx] = hn;
    }
  }
}

__global__ __launch_bounds__(512) void lstm_kernel(const _Float16* __restrict__ feats,
                            const _Float16* __restrict__ wih0, const _Float16* __restrict__ whh0,
                            const _Float16* __restrict__ wih1, const _Float16* __restrict__ whh1,
                            const float* __restrict__ fc1w, const float* __restrict__ fc2w,
                            const int* __restrict__ lengths, float* __restrict__ out) {
  __shared__ alignas(32) _Float16 h0[BB * HH];
  __shared__ alignas(32) _Float16 h1[BB * HH];
  __shared__ float c0[BB * HH];
  __shared__ float c1[BB * HH];
  __shared__ float h1f[BB * HH];
  __shared__ alignas(32) float gates[512 * BB];
  __shared__ float fc1o[BB * 10];
  for (int i = threadIdx.x; i < BB * HH; i += blockDim.x) {
    h0[i] = (_Float16)0.0f; h1[i] = (_Float16)0.0f;
    c0[i] = 0.0f; c1[i] = 0.0f; h1f[i] = 0.0f;
  }
  __syncthreads();
  for (int t = 0; t < TT; ++t) {
    lstm_gemm(feats + (size_t)t * BB * 128, h0, wih0, whh0, gates);
    __syncthreads();
    lstm_act(gates, c0, h0, nullptr, lengths, t);
    __syncthreads();
    // layer 1 consumes layer 0's fresh h directly from LDS (pipelined)
    lstm_gemm(h0, h1, wih1, whh1, gates);
    __syncthreads();
    lstm_act(gates, c1, h1, h1f, lengths, t);
    __syncthreads();
  }
  if (threadIdx.x < 320) {
    int b = threadIdx.x / 10, o = threadIdx.x % 10;
    float s = 0.0f;
    for (int k = 0; k < HH; ++k) s += fc1w[o * HH + k] * h1f[b * HH + k];
    fc1o[b * 10 + o] = LEAKY(s);
  }
  __syncthreads();
  if (threadIdx.x < 64) {
    int b = threadIdx.x >> 1, o = threadIdx.x & 1;
    float s = 0.0f;
    for (int k = 0; k < 10; ++k) s += fc2w[o * 10 + k] * fc1o[b * 10 + k];
    out[b * 2 + o] = LEAKY(s);
  }
}

extern "C" void kernel_launch(void* const* d_in, const int* in_sizes, int n_in,
                              void* d_out, int out_size, void* d_ws, size_t ws_size,
                              hipStream_t stream) {
  (void)in_sizes; (void)n_in; (void)out_size; (void)ws_size;
  const float* X       = (const float*)d_in[0];
  const int*   lengths = (const int*)d_in[1];
  const float* c1w     = (const float*)d_in[2];
  const float* c2w     = (const float*)d_in[3];
  const float* c3w     = (const float*)d_in[4];
  const float* linw    = (const float*)d_in[5];
  const float* wih0    = (const float*)d_in[6];
  const float* whh0    = (const float*)d_in[7];
  const float* wih1    = (const float*)d_in[8];
  const float* whh1    = (const float*)d_in[9];
  const float* fc1w    = (const float*)d_in[10];
  const float* fc2w    = (const float*)d_in[11];
  float* out = (float*)d_out;

  _Float16* wp = (_Float16*)d_ws;
  _Float16* w2h   = wp; wp += 32 * 640;
  _Float16* w3h   = wp; wp += 32 * 160;
  _Float16* linh  = wp; wp += DCC * FLATN;
  _Float16* wih0h = wp; wp += 512 * 128;
  _Float16* whh0h = wp; wp += 512 * 128;
  _Float16* wih1h = wp; wp += 512 * 128;
  _Float16* whh1h = wp; wp += 512 * 128;
  _Float16* feats = wp; wp += (size_t)TT * BB * 128;
  _Float16* a2    = wp; wp += (size_t)NBT * 32 * P2;
  _Float16* a3    = wp; wp += (size_t)NBT * FLATN;

  prep_kernel<<<512, 256, 0, stream>>>(c2w, c3w, linw, wih0, whh0, wih1, whh1,
                                       w2h, w3h, linh, wih0h, whh0h, wih1h, whh1h, feats);
  conv12_kernel<<<dim3(NBT, 6), 128, 0, stream>>>(X, c1w, w2h, a2);
  conv3_kernel<<<NBT, 256, 0, stream>>>(a2, w3h, a3);
  linear_kernel<<<dim3(128, 7), 32, 0, stream>>>(a3, linh, feats);
  lstm_kernel<<<1, 512, 0, stream>>>(feats, wih0h, whh0h, wih1h, whh1h,
                                     fc1w, fc2w, lengths, out);
}
